// GINENN_46583215292858
// MI455X (gfx1250) — matrix operand order repaired and checked by reading, not compiled, not run on hardware
//
#include <hip/hip_runtime.h>

#define NN 50000
#define NE 300000
#define NG 512
#define H 256
#define NL 5
#define BN_EPS 1e-5f

typedef __attribute__((ext_vector_type(16))) __bf16 v16bf;
typedef __attribute__((ext_vector_type(8)))  float v8f;
typedef __attribute__((ext_vector_type(8)))  unsigned int v8u;

// ---------- helpers ----------

__device__ __forceinline__ unsigned short f2bf(float f) {
  unsigned int u = __builtin_bit_cast(unsigned int, f);
  u += 0x7FFFu + ((u >> 16) & 1u);   // round-to-nearest-even
  return (unsigned short)(u >> 16);
}

// Packed A-operand layout for V_WMMA_F32_16X16X32_BF16 (ISA 7.12.2):
// A 16x32 bf16; lane = (remK/8)*16 + row%16 ; vgpr v = (k%32/16)*4 + (k%8)/2
// Returns index (in uint32 units, each uint holds bf16 pair k,k+1) for even k.
__device__ __forceinline__ size_t apack_idx(int row, int k) {
  int et = row >> 4, m = row & 15;
  int kt = k >> 5, kk = k & 31;
  int group = kk >> 4, rem = kk & 15;
  int half = rem >> 3, sub = (rem & 7) >> 1;
  int v = group * 4 + sub;
  int lane = half * 16 + m;
  return ((((size_t)et * (H / 32) + kt) * 32) + lane) * 8 + v;
}

__device__ __forceinline__ void atomicMaxF(float* addr, float val) {
  unsigned int* ua = (unsigned int*)addr;
  unsigned int old = *ua;
  while (__builtin_bit_cast(float, old) < val) {
    unsigned int assumed = old;
    old = atomicCAS(ua, assumed, __builtin_bit_cast(unsigned int, val));
    if (old == assumed) break;
  }
}

// ---------- utility kernels ----------

__global__ void fill_f32(float* p, float v, int n) {
  int i = blockIdx.x * 256 + threadIdx.x;
  if (i < n) p[i] = v;
}

// Pack lin_edge_w (5) + mlp_w (5) into WMMA B-operand bf16 layout.
// B 32x16 bf16 per 16x16x32 tile: lane -> col = lane%16, k = (v/4)*16 + (lane/16)*8 + (v%4)*2.
__global__ void pack_weights(const float* __restrict__ lin_w,
                             const float* __restrict__ mlp_w,
                             unsigned int* __restrict__ Wp) {
  int t = blockIdx.x * 256 + threadIdx.x;    // 10 * 32768 uints total
  if (t >= 10 * 32768) return;
  int mat = t >> 15;
  int r = t & 32767;                          // ((nt*8+kt)*32+lane)*8+v
  int v = r & 7, lane = (r >> 3) & 31, kt = (r >> 8) & 7, nt = r >> 11;
  int k = kt * 32 + (v >> 2) * 16 + (lane >> 4) * 8 + (v & 3) * 2;
  int n = nt * 16 + (lane & 15);
  const float* W = (mat < 5) ? (lin_w + (size_t)mat * H * H)
                             : (mlp_w + (size_t)(mat - 5) * H * H);
  unsigned int lo = f2bf(W[(size_t)k * H + n]);
  unsigned int hi = f2bf(W[(size_t)(k + 1) * H + n]);
  Wp[t] = lo | (hi << 16);
}

// AtomEncoder: h[n,c] = sum_{j<9} atom_emb[j, x[n,j], c]
__global__ void atom_enc(const int* __restrict__ x, const float* __restrict__ emb,
                         float* __restrict__ h) {
  int i = blockIdx.x * 256 + threadIdx.x;
  if (i >= NN * H) return;
  int node = i >> 8, c = i & 255;
  float s = 0.f;
#pragma unroll
  for (int j = 0; j < 9; ++j)
    s += emb[((size_t)j * 64 + x[node * 9 + j]) * H + c];
  h[i] = s;
}

// BondEncoder, written directly in WMMA-A packed bf16 layout (e is reused 5x as GEMM A).
__global__ void bond_enc(const int* __restrict__ ea, const float* __restrict__ emb,
                         unsigned int* __restrict__ eA) {
  int t = blockIdx.x * 256 + threadIdx.x;    // NE * H/2
  if (t >= NE * (H / 2)) return;
  int ed = t >> 7;
  int kp = (t & 127) << 1;
  float s0 = 0.f, s1 = 0.f;
#pragma unroll
  for (int c = 0; c < 3; ++c) {
    const float* base = emb + ((size_t)c * 16 + ea[ed * 3 + c]) * H;
    s0 += base[kp];
    s1 += base[kp + 1];
  }
  eA[apack_idx(ed, kp)] = (unsigned int)f2bf(s0) | ((unsigned int)f2bf(s1) << 16);
}

// z = (1+eps_l)*h + agg, packed to WMMA-A bf16 layout for the node GEMM.
__global__ void zpack(const float* __restrict__ h, const float* __restrict__ agg,
                      const float* __restrict__ eps, int l,
                      unsigned int* __restrict__ zA) {
  int t = blockIdx.x * 256 + threadIdx.x;    // NN * H/2
  if (t >= NN * (H / 2)) return;
  float scale = 1.0f + eps[l];
  int node = t >> 7;
  int kp = (t & 127) << 1;
  size_t base = (size_t)node * H + kp;
  float z0 = scale * h[base] + agg[base];
  float z1 = scale * h[base + 1] + agg[base + 1];
  zA[apack_idx(node, kp)] = (unsigned int)f2bf(z0) | ((unsigned int)f2bf(z1) << 16);
}

// ---------- WMMA GEMMs ----------

// Edge GEMM fused with GINE message + scatter:
//   e_l = e @ W + b ; m = relu(h[src] + e_l) ; agg[dst] += m
// One wave computes a 16(edge) x 64(col) tile; K = 256 in 8 steps of 32.
__global__ __launch_bounds__(256) void edge_gemm_scatter(
    const unsigned int* __restrict__ eA, const unsigned int* __restrict__ Wp,
    const float* __restrict__ bias, const int* __restrict__ ei,
    const float* __restrict__ h, float* __restrict__ agg) {
  int lane = threadIdx.x & 31, wave = threadIdx.x >> 5;
  int tile = blockIdx.x * 8 + wave;
  const int ntiles = (NE / 16) * 4;
  if (tile >= ntiles) return;
  int et = tile >> 2, n64 = tile & 3;
  v8f acc[4] = {};
  for (int kt = 0; kt < 8; ++kt) {
    v8u au = *(const v8u*)&eA[(((size_t)et * 8 + kt) * 32 + lane) * 8];
    v16bf a = __builtin_bit_cast(v16bf, au);
#pragma unroll
    for (int j = 0; j < 4; ++j) {
      v8u bu = *(const v8u*)&Wp[(((size_t)(n64 * 4 + j) * 8 + kt) * 256) + lane * 8];
      acc[j] = __builtin_amdgcn_wmma_f32_16x16x32_bf16(
          false, a, false, __builtin_bit_cast(v16bf, bu), (short)0, acc[j],
          false, false);
    }
  }
  const int* src = ei;
  const int* dst = ei + NE;
  int half = lane >> 4, cIn = lane & 15;
#pragma unroll
  for (int j = 0; j < 4; ++j) {
    int col = (n64 * 4 + j) * 16 + cIn;
    float bb = bias[col];
#pragma unroll
    for (int v = 0; v < 8; ++v) {     // D layout: lane row = v + 8*(lane/16)
      int edge = et * 16 + v + half * 8;
      int s = src[edge], d = dst[edge];
      float val = acc[j][v] + bb + h[(size_t)s * H + col];
      val = fmaxf(val, 0.f);
      atomicAdd(&agg[(size_t)d * H + col], val);
    }
  }
}

// Node MLP GEMM: zmid = z @ mlp_w + b (dense output, BN handled separately).
__global__ __launch_bounds__(256) void node_gemm(
    const unsigned int* __restrict__ zA, const unsigned int* __restrict__ Wp,
    const float* __restrict__ bias, float* __restrict__ zmid) {
  int lane = threadIdx.x & 31, wave = threadIdx.x >> 5;
  int tile = blockIdx.x * 8 + wave;
  const int ntiles = (NN / 16) * 4;
  if (tile >= ntiles) return;
  int mt = tile >> 2, n64 = tile & 3;
  v8f acc[4] = {};
  for (int kt = 0; kt < 8; ++kt) {
    v8u au = *(const v8u*)&zA[(((size_t)mt * 8 + kt) * 32 + lane) * 8];
    v16bf a = __builtin_bit_cast(v16bf, au);
#pragma unroll
    for (int j = 0; j < 4; ++j) {
      v8u bu = *(const v8u*)&Wp[(((size_t)(n64 * 4 + j) * 8 + kt) * 256) + lane * 8];
      acc[j] = __builtin_amdgcn_wmma_f32_16x16x32_bf16(
          false, a, false, __builtin_bit_cast(v16bf, bu), (short)0, acc[j],
          false, false);
    }
  }
  int half = lane >> 4, cIn = lane & 15;
#pragma unroll
  for (int j = 0; j < 4; ++j) {
    int col = (n64 * 4 + j) * 16 + cIn;
    float bb = bias[col];
#pragma unroll
    for (int v = 0; v < 8; ++v) {
      int row = mt * 16 + v + half * 8;
      zmid[(size_t)row * H + col] = acc[j][v] + bb;
    }
  }
}

// ---------- BatchNorm ----------

__global__ void bn_stats(const float* __restrict__ X, float* __restrict__ bsum,
                         float* __restrict__ bsq) {
  int c = threadIdx.x;                        // blockDim = H = 256
  int r0 = blockIdx.x * 128;
  int r1 = r0 + 128; if (r1 > NN) r1 = NN;
  float s = 0.f, q = 0.f;
  for (int r = r0; r < r1; ++r) {
    float v = X[(size_t)r * H + c];
    s += v; q += v * v;
  }
  atomicAdd(&bsum[c], s);
  atomicAdd(&bsq[c], q);
}

// t = relu( relu(BN(zmid)) + h )
__global__ void bn_apply1(const float* __restrict__ zmid, const float* __restrict__ h,
                          const float* __restrict__ bsum, const float* __restrict__ bsq,
                          const float* __restrict__ g, const float* __restrict__ b,
                          float* __restrict__ tbuf) {
  int i = blockIdx.x * 256 + threadIdx.x;
  if (i >= NN * H) return;
  int c = i & 255;
  const float invN = 1.0f / (float)NN;
  float mu = bsum[c] * invN;
  float var = bsq[c] * invN - mu * mu;
  float z = (zmid[i] - mu) * rsqrtf(var + BN_EPS) * g[c] + b[c];
  z = fmaxf(z, 0.f);
  tbuf[i] = fmaxf(z + h[i], 0.f);
}

// h = BN(t)
__global__ void bn_apply2(const float* __restrict__ tbuf,
                          const float* __restrict__ bsum, const float* __restrict__ bsq,
                          const float* __restrict__ g, const float* __restrict__ b,
                          float* __restrict__ h) {
  int i = blockIdx.x * 256 + threadIdx.x;
  if (i >= NN * H) return;
  int c = i & 255;
  const float invN = 1.0f / (float)NN;
  float mu = bsum[c] * invN;
  float var = bsq[c] * invN - mu * mu;
  h[i] = (tbuf[i] - mu) * rsqrtf(var + BN_EPS) * g[c] + b[c];
}

// ---------- pooling + classifier ----------

__global__ void pool_accum(const float* __restrict__ h, const int* __restrict__ batch,
                           float* __restrict__ gsum, float* __restrict__ gmax,
                           int* __restrict__ gcnt) {
  int i = blockIdx.x * 256 + threadIdx.x;
  if (i >= NN * H) return;
  int node = i >> 8, c = i & 255;
  int g = batch[node];
  float v = h[i];
  atomicAdd(&gsum[(size_t)g * H + c], v);
  atomicMaxF(&gmax[(size_t)g * H + c], v);
  if (c == 0) atomicAdd(&gcnt[g], 1);
}

__global__ __launch_bounds__(256) void classifier(
    const float* __restrict__ gsum, const float* __restrict__ gmax,
    const int* __restrict__ gcnt, const float* __restrict__ W1,
    const float* __restrict__ b1, const float* __restrict__ W2,
    const float* __restrict__ b2, float* __restrict__ out) {
  __shared__ float mean_s[H];
  __shared__ float max_s[H];
  __shared__ float red[256];
  int g = blockIdx.x, j = threadIdx.x;
  float cnt = fmaxf((float)gcnt[g], 1.0f);
  mean_s[j] = gsum[(size_t)g * H + j] / cnt;
  float mv = gmax[(size_t)g * H + j];
  max_s[j] = ((__builtin_bit_cast(unsigned int, mv) & 0x7F800000u) != 0x7F800000u)
                 ? mv : 0.0f;                 // isfinite ? mv : 0
  __syncthreads();
  float acc = b1[j];
  for (int k = 0; k < H; ++k) acc += mean_s[k] * W1[(size_t)k * H + j];
  for (int k = 0; k < H; ++k) acc += max_s[k] * W1[(size_t)(H + k) * H + j];
  red[j] = fmaxf(acc, 0.f) * W2[j];
  __syncthreads();
  for (int s = 128; s > 0; s >>= 1) {
    if (j < s) red[j] += red[j + s];
    __syncthreads();
  }
  if (j == 0) out[g] = red[0] + b2[0];
}

// ---------- host ----------

extern "C" void kernel_launch(void* const* d_in, const int* in_sizes, int n_in,
                              void* d_out, int out_size, void* d_ws, size_t ws_size,
                              hipStream_t stream) {
  (void)in_sizes; (void)n_in; (void)out_size; (void)ws_size;
  const int*   x        = (const int*)  d_in[0];
  const int*   ei       = (const int*)  d_in[1];
  const int*   batch    = (const int*)  d_in[2];
  const int*   eattr    = (const int*)  d_in[3];
  const float* atom_emb = (const float*)d_in[4];
  const float* bond_emb = (const float*)d_in[5];
  const float* eps      = (const float*)d_in[6];
  const float* lin_w    = (const float*)d_in[7];
  const float* lin_b    = (const float*)d_in[8];
  const float* mlp_w    = (const float*)d_in[9];
  const float* mlp_b    = (const float*)d_in[10];
  const float* mbn_g    = (const float*)d_in[11];
  const float* mbn_b    = (const float*)d_in[12];
  const float* bn_g     = (const float*)d_in[13];
  const float* bn_b     = (const float*)d_in[14];
  const float* w1       = (const float*)d_in[15];
  const float* b1       = (const float*)d_in[16];
  const float* w2       = (const float*)d_in[17];
  const float* b2       = (const float*)d_in[18];
  float* out = (float*)d_out;

  char* ws = (char*)d_ws;
  size_t off = 0;
  auto take = [&](size_t bytes) -> char* {
    char* p = ws + off;
    off = (off + bytes + 255) & ~(size_t)255;
    return p;
  };
  float*        h    = (float*)take((size_t)NN * H * 4);
  float*        agg  = (float*)take((size_t)NN * H * 4);   // also reused as tbuf
  float*        zmid = (float*)take((size_t)NN * H * 4);
  unsigned int* eA   = (unsigned int*)take((size_t)NE * H * 2);
  unsigned int* zA   = (unsigned int*)take((size_t)NN * H * 2);
  unsigned int* Wp   = (unsigned int*)take((size_t)10 * H * H * 2);
  float*        bsum = (float*)take(H * 4);
  float*        bsq  = (float*)take(H * 4);
  float*        gsum = (float*)take((size_t)NG * H * 4);
  float*        gmax = (float*)take((size_t)NG * H * 4);
  int*          gcnt = (int*)  take((size_t)NG * 4);

  const int NH = NN * H;
  auto gB = [](int n) { return (n + 255) / 256; };

  pack_weights<<<gB(10 * 32768), 256, 0, stream>>>(lin_w, mlp_w, Wp);
  atom_enc<<<gB(NH), 256, 0, stream>>>(x, atom_emb, h);
  bond_enc<<<gB(NE * (H / 2)), 256, 0, stream>>>(eattr, bond_emb, eA);

  for (int l = 0; l < NL; ++l) {
    fill_f32<<<gB(NH), 256, 0, stream>>>(agg, 0.f, NH);
    edge_gemm_scatter<<<((NE / 16) * 4 + 7) / 8, 256, 0, stream>>>(
        eA, Wp + (size_t)l * 32768, lin_b + (size_t)l * H, ei, h, agg);
    zpack<<<gB(NN * (H / 2)), 256, 0, stream>>>(h, agg, eps, l, zA);
    node_gemm<<<((NN / 16) * 4 + 7) / 8, 256, 0, stream>>>(
        zA, Wp + (size_t)(5 + l) * 32768, mlp_b + (size_t)l * H, zmid);

    fill_f32<<<1, 256, 0, stream>>>(bsum, 0.f, H);
    fill_f32<<<1, 256, 0, stream>>>(bsq, 0.f, H);
    bn_stats<<<(NN + 127) / 128, 256, 0, stream>>>(zmid, bsum, bsq);
    bn_apply1<<<gB(NH), 256, 0, stream>>>(zmid, h, bsum, bsq,
                                          mbn_g + (size_t)l * H,
                                          mbn_b + (size_t)l * H, agg);
    fill_f32<<<1, 256, 0, stream>>>(bsum, 0.f, H);
    fill_f32<<<1, 256, 0, stream>>>(bsq, 0.f, H);
    bn_stats<<<(NN + 127) / 128, 256, 0, stream>>>(agg, bsum, bsq);
    bn_apply2<<<gB(NH), 256, 0, stream>>>(agg, bsum, bsq,
                                          bn_g + (size_t)l * H,
                                          bn_b + (size_t)l * H, h);
  }

  fill_f32<<<gB(NG * H), 256, 0, stream>>>(gsum, 0.f, NG * H);
  fill_f32<<<gB(NG * H), 256, 0, stream>>>(gmax, -__builtin_huge_valf(), NG * H);
  fill_f32<<<gB(NG), 256, 0, stream>>>((float*)gcnt, 0.f, NG);
  pool_accum<<<gB(NH), 256, 0, stream>>>(h, batch, gsum, gmax, gcnt);
  classifier<<<NG, 256, 0, stream>>>(gsum, gmax, gcnt, w1, b1, w2, b2, out);
}
